// MultiHeadAttention_74225624809977
// MI455X (gfx1250) — compile-verified, matrix operand order checked by reading, not run on hardware
//
#include <hip/hip_runtime.h>
#include <hip/hip_bf16.h>
#include <stdint.h>

// ---------------------------------------------------------------------------
// MI455X (gfx1250, wave32) fused causal MHA forward:
//   0) pre-pass: X f32->bf16; W_qkv/W_0 f32 -> bf16 transposed [N][K]
//   1) QKV projection GEMM (bf16 WMMA, 128x256 block / 64x64 wave tile)
//   2) flash-attention (bf16 WMMA, f32 accum, online softmax)
//   3) output projection GEMM (f32 out + bias)
// GEMM LDS staging uses GLOBAL_LOAD_ASYNC_TO_LDS_B128 when available.
// ---------------------------------------------------------------------------

typedef __bf16 bf16;
typedef bf16  bf16x4  __attribute__((ext_vector_type(4)));
typedef bf16  bf16x8  __attribute__((ext_vector_type(8)));
typedef bf16  bf16x16 __attribute__((ext_vector_type(16)));
typedef float floatx8 __attribute__((ext_vector_type(8)));
typedef int   intx4   __attribute__((ext_vector_type(4)));

typedef __attribute__((address_space(1))) intx4 g_intx4;  // global int4
typedef __attribute__((address_space(3))) intx4 l_intx4;  // LDS int4

#if defined(__has_builtin)
#if __has_builtin(__builtin_amdgcn_global_load_async_to_lds_b128) && \
    __has_builtin(__builtin_amdgcn_s_wait_asynccnt)
#define USE_ASYNC_LDS 1
#endif
#endif

#define WMMA_BF16(A, B, C)                                                    \
  __builtin_amdgcn_wmma_f32_16x16x32_bf16(false, (A), false, (B), (short)0,   \
                                          (C), false, false)

// Copy 16 bytes (8 x bf16) global -> LDS; async (ASYNCcnt) when available.
static __device__ __forceinline__ void cp16_g2s(const bf16* g, bf16* l) {
#ifdef USE_ASYNC_LDS
  __builtin_amdgcn_global_load_async_to_lds_b128((g_intx4*)g, (l_intx4*)l,
                                                 0, 0);
#else
  *(bf16x8*)l = *(const bf16x8*)g;
#endif
}
static __device__ __forceinline__ void cp_join() {
#ifdef USE_ASYNC_LDS
  __builtin_amdgcn_s_wait_asynccnt(0);
#endif
}

// Fragment element e (0..15) maps to K = 8*(lane/16) + e       for e in 0..7
//                               K = 16 + 8*(lane/16) + (e-8)   for e in 8..15
static __device__ __forceinline__ bf16x16 frag_ld(const bf16* base, int k0) {
  bf16x8 lo = *(const bf16x8*)(base + k0);
  bf16x8 hi = *(const bf16x8*)(base + k0 + 16);
  bf16x16 r;
#pragma unroll
  for (int i = 0; i < 8; ++i) { r[i] = lo[i]; r[i + 8] = hi[i]; }
  return r;
}

static __device__ __forceinline__ floatx8 fzero8() {
  floatx8 z = {0.f, 0.f, 0.f, 0.f, 0.f, 0.f, 0.f, 0.f};
  return z;
}

// ---------------------------------------------------------------------------
// Pre-pass 0a: elementwise f32 -> bf16 (X).
// ---------------------------------------------------------------------------
__global__ __launch_bounds__(256) void mha_cvt_bf16(
    const float* __restrict__ src, bf16* __restrict__ dst) {
  const size_t i = (size_t)blockIdx.x * 256 + threadIdx.x;  // float4 index
  const float4 v = *(const float4*)(src + i * 4);
  bf16x4 b;
  b[0] = (bf16)v.x; b[1] = (bf16)v.y; b[2] = (bf16)v.z; b[3] = (bf16)v.w;
  *(bf16x4*)(dst + i * 4) = b;
}

// ---------------------------------------------------------------------------
// Pre-pass 0b: transpose+convert W[K][N] f32 -> Wt[N][K] bf16 (32x32 tiles).
// ---------------------------------------------------------------------------
__global__ __launch_bounds__(256) void mha_transpose_bf16(
    const float* __restrict__ src, bf16* __restrict__ dst, int K, int N) {
  __shared__ float t[32][33];
  const int tid = threadIdx.x;
  const int lx = tid & 31, ly = tid >> 5;  // ly in 0..7
  const int bx = blockIdx.x * 32;          // n base
  const int by = blockIdx.y * 32;          // k base
#pragma unroll
  for (int r = 0; r < 4; ++r) {
    const int kr = ly + r * 8;
    t[kr][lx] = src[(size_t)(by + kr) * N + bx + lx];
  }
  __syncthreads();
#pragma unroll
  for (int r = 0; r < 4; ++r) {
    const int nr = ly + r * 8;
    dst[(size_t)(bx + nr) * K + by + lx] = (bf16)t[lx][nr];
  }
}

// ---------------------------------------------------------------------------
// Kernel 1: QKV projection. C[4096,3072] = Xb[4096,1024] @ Wt^T + b
// Block tile 128x256, 8 waves (2x4), wave tile 64x64 (4x4 WMMA tiles).
// A (Xb) and B (Wt, already [n][k]) staged as pure bf16 b128 copies.
// Epilogue scatters into Q/K/V [B,H,S,64] bf16; Q scaled by 1/sqrt(64).
// ---------------------------------------------------------------------------
__global__ __launch_bounds__(256) void mha_qkv_gemm(
    const bf16* __restrict__ Xb, const bf16* __restrict__ Wt,
    const float* __restrict__ bqkv, bf16* __restrict__ Qb,
    bf16* __restrict__ Kb, bf16* __restrict__ Vb) {
  constexpr int LDK = 40;
  __shared__ bf16 As[128 * LDK];  // [m][k]
  __shared__ bf16 Bs[256 * LDK];  // [n][k]

  const int tid = threadIdx.x;
  const int wave = tid >> 5, lane = tid & 31;
  const int lane16 = lane & 15, half = lane >> 4;
  const int k0 = half * 8;
  const int wm = wave & 1, wn = wave >> 1;
  const int mbase = blockIdx.y * 128;
  const int nbase = blockIdx.x * 256;

  floatx8 acc[4][4];
#pragma unroll
  for (int mt = 0; mt < 4; ++mt)
#pragma unroll
    for (int nt = 0; nt < 4; ++nt) acc[mt][nt] = fzero8();

  const bf16* a_base[4];
  const bf16* b_base[4];
#pragma unroll
  for (int mt = 0; mt < 4; ++mt)
    a_base[mt] = &As[(wm * 64 + mt * 16 + lane16) * LDK];
#pragma unroll
  for (int nt = 0; nt < 4; ++nt)
    b_base[nt] = &Bs[(wn * 64 + nt * 16 + lane16) * LDK];

  for (int it = 0; it < 32; ++it) {
    const int kb = it * 32;
    __syncthreads();
    // Stage A: 128x32 bf16 copy (512 x 16B chunks, 2 per thread)
#pragma unroll
    for (int i = 0; i < 2; ++i) {
      const int id = tid + i * 256;
      const int row = id >> 2, c = id & 3;
      cp16_g2s(Xb + (size_t)(mbase + row) * 1024 + kb + c * 8,
               &As[row * LDK + c * 8]);
    }
    // Stage B: 256x32 bf16 copy (1024 x 16B chunks, 4 per thread)
#pragma unroll
    for (int i = 0; i < 4; ++i) {
      const int id = tid + i * 256;
      const int n = id >> 2, c = id & 3;
      cp16_g2s(Wt + (size_t)(nbase + n) * 1024 + kb + c * 8,
               &Bs[n * LDK + c * 8]);
    }
    cp_join();
    __syncthreads();

    bf16x16 bfm[4];
#pragma unroll
    for (int nt = 0; nt < 4; ++nt) bfm[nt] = frag_ld(b_base[nt], k0);
#pragma unroll
    for (int mt = 0; mt < 4; ++mt) {
      bf16x16 af = frag_ld(a_base[mt], k0);
#pragma unroll
      for (int nt = 0; nt < 4; ++nt)
        acc[mt][nt] = WMMA_BF16(af, bfm[nt], acc[mt][nt]);
    }
  }

  // Epilogue: bias add + scatter into head-major Q/K/V (bf16).
  const int t_sec = nbase >> 10;  // constant per block: 0=Q,1=K,2=V
  bf16* dst = (t_sec == 0) ? Qb : ((t_sec == 1) ? Kb : Vb);
  const float scale = (t_sec == 0) ? 0.125f : 1.0f;  // 1/sqrt(HD) into Q
#pragma unroll
  for (int mt = 0; mt < 4; ++mt) {
#pragma unroll
    for (int nt = 0; nt < 4; ++nt) {
      const int ng = nbase + wn * 64 + nt * 16 + lane16;
      const float bias = bqkv[ng];
      const int d = ng & 1023;
      const int hh = d >> 6, dim = d & 63;
#pragma unroll
      for (int r = 0; r < 8; ++r) {
        const int mg = mbase + wm * 64 + mt * 16 + r + half * 8;
        const int bb = mg >> 11, s = mg & 2047;
        const float v = (acc[mt][nt][r] + bias) * scale;
        dst[(((size_t)(bb * 16 + hh) * 2048 + s) << 6) + dim] = (bf16)v;
      }
    }
  }
}

// ---------------------------------------------------------------------------
// Kernel 2: causal flash-attention per (b,h, 128-query tile).
// 8 waves x 16 query rows; key tiles of 32; online softmax; f32 accum.
// ---------------------------------------------------------------------------
__global__ __launch_bounds__(256) void mha_flash_attn(
    const bf16* __restrict__ Qb, const bf16* __restrict__ Kb,
    const bf16* __restrict__ Vb, bf16* __restrict__ Oattn) {
  constexpr int LDK = 40;
  __shared__ bf16 Vt[64 * LDK];       // V tile transposed: [hd][key]
  __shared__ bf16 Pws[8 * 16 * LDK];  // per-wave P scratch: [row][key]

  const int tid = threadIdx.x;
  const int wave = tid >> 5, lane = tid & 31;
  const int lane16 = lane & 15, half = lane >> 4;
  const int k0 = half * 8;
  const int bh = blockIdx.y;
  const int qt = blockIdx.x;
  const size_t ho = (size_t)bh * (2048 * 64);
  const bf16* qp = Qb + ho;
  const bf16* kp = Kb + ho;
  const bf16* vp = Vb + ho;
  const int qrow0 = qt * 128 + wave * 16;

  bf16x16 qf[2];
#pragma unroll
  for (int c = 0; c < 2; ++c)
    qf[c] = frag_ld(qp + (size_t)(qrow0 + lane16) * 64 + c * 32, k0);

  floatx8 accO[4];
#pragma unroll
  for (int i = 0; i < 4; ++i) accO[i] = fzero8();
  float mrow[8], lrow[8];
#pragma unroll
  for (int r = 0; r < 8; ++r) { mrow[r] = -1e30f; lrow[r] = 0.f; }

  const int nkt = qt * 4 + 4;
  for (int kt = 0; kt < nkt; ++kt) {
    const int kb = kt * 32;
    __syncthreads();
    {  // cooperative transposed stage of V tile: Vt[hd][key]
      const int key = tid >> 3;
      const int h0 = (tid & 7) * 8;
      bf16x8 vv = *(const bf16x8*)(vp + (size_t)(kb + key) * 64 + h0);
#pragma unroll
      for (int j = 0; j < 8; ++j) Vt[(h0 + j) * LDK + key] = vv[j];
    }
    __syncthreads();

    if (kt + 1 < nkt) {  // prefetch next K/V key tile
      const int nk = kb + 32 + lane;
      __builtin_prefetch(kp + (size_t)nk * 64, 0, 0);
      __builtin_prefetch(vp + (size_t)nk * 64, 0, 0);
    }

    // scores S = Q @ K^T  (K fragments straight from global: [key][hd])
    floatx8 sacc[2];
    sacc[0] = fzero8(); sacc[1] = fzero8();
#pragma unroll
    for (int nt = 0; nt < 2; ++nt) {
      const bf16* kbp = kp + (size_t)(kb + nt * 16 + lane16) * 64;
#pragma unroll
      for (int c = 0; c < 2; ++c) {
        bf16x16 kf = frag_ld(kbp + c * 32, k0);
        sacc[nt] = WMMA_BF16(qf[c], kf, sacc[nt]);
      }
    }

    // causal mask
#pragma unroll
    for (int nt = 0; nt < 2; ++nt)
#pragma unroll
      for (int r = 0; r < 8; ++r) {
        const int kcol = kb + nt * 16 + lane16;
        const int qrow = qrow0 + r + half * 8;
        if (kcol > qrow) sacc[nt][r] = -1e30f;
      }

    // online softmax (row lives across 16 lanes of each half-wave)
#pragma unroll
    for (int r = 0; r < 8; ++r) {
      float rm = fmaxf(sacc[0][r], sacc[1][r]);
      rm = fmaxf(rm, __shfl_xor(rm, 1));
      rm = fmaxf(rm, __shfl_xor(rm, 2));
      rm = fmaxf(rm, __shfl_xor(rm, 4));
      rm = fmaxf(rm, __shfl_xor(rm, 8));
      const float mnew = fmaxf(mrow[r], rm);
      const float fac = __expf(mrow[r] - mnew);
      const float p0 = __expf(sacc[0][r] - mnew);
      const float p1 = __expf(sacc[1][r] - mnew);
      sacc[0][r] = p0; sacc[1][r] = p1;
      float rs = p0 + p1;
      rs += __shfl_xor(rs, 1);
      rs += __shfl_xor(rs, 2);
      rs += __shfl_xor(rs, 4);
      rs += __shfl_xor(rs, 8);
      lrow[r] = lrow[r] * fac + rs;
      mrow[r] = mnew;
#pragma unroll
      for (int nt4 = 0; nt4 < 4; ++nt4) accO[nt4][r] *= fac;
    }

    // reshape P: C-layout -> A-fragment via per-wave LDS round-trip
    bf16* pw = &Pws[wave * 16 * LDK];
#pragma unroll
    for (int nt = 0; nt < 2; ++nt)
#pragma unroll
      for (int r = 0; r < 8; ++r)
        pw[(r + half * 8) * LDK + nt * 16 + lane16] = (bf16)sacc[nt][r];
    bf16x16 pa = frag_ld(pw + lane16 * LDK, k0);

    // O += P @ V
#pragma unroll
    for (int nt4 = 0; nt4 < 4; ++nt4) {
      bf16x16 vf = frag_ld(&Vt[(nt4 * 16 + lane16) * LDK], k0);
      accO[nt4] = WMMA_BF16(pa, vf, accO[nt4]);
    }
  }

  // finalize: O / l, store to attention output [B,S,H*HD] bf16
  const int bidx = bh >> 4, hh = bh & 15;
#pragma unroll
  for (int nt4 = 0; nt4 < 4; ++nt4)
#pragma unroll
    for (int r = 0; r < 8; ++r) {
      const int mg = qrow0 + r + half * 8;
      const int col = hh * 64 + nt4 * 16 + lane16;
      const float v = accO[nt4][r] / lrow[r];
      Oattn[((size_t)(bidx * 2048 + mg)) * 1024 + col] = (bf16)v;
    }
}

// ---------------------------------------------------------------------------
// Kernel 3: output projection. Out[4096,1024] = A_bf16 @ Wt0^T + b0 (f32 out)
// Block tile 128x256, wave tile 64x64.
// ---------------------------------------------------------------------------
__global__ __launch_bounds__(256) void mha_out_gemm(
    const bf16* __restrict__ Ain, const bf16* __restrict__ Wt0,
    const float* __restrict__ b0, float* __restrict__ Out) {
  constexpr int LDK = 40;
  __shared__ bf16 As[128 * LDK];
  __shared__ bf16 Bs[256 * LDK];

  const int tid = threadIdx.x;
  const int wave = tid >> 5, lane = tid & 31;
  const int lane16 = lane & 15, half = lane >> 4;
  const int k0 = half * 8;
  const int wm = wave & 1, wn = wave >> 1;
  const int mbase = blockIdx.y * 128;
  const int nbase = blockIdx.x * 256;

  floatx8 acc[4][4];
#pragma unroll
  for (int mt = 0; mt < 4; ++mt)
#pragma unroll
    for (int nt = 0; nt < 4; ++nt) acc[mt][nt] = fzero8();

  const bf16* a_base[4];
  const bf16* b_base[4];
#pragma unroll
  for (int mt = 0; mt < 4; ++mt)
    a_base[mt] = &As[(wm * 64 + mt * 16 + lane16) * LDK];
#pragma unroll
  for (int nt = 0; nt < 4; ++nt)
    b_base[nt] = &Bs[(wn * 64 + nt * 16 + lane16) * LDK];

  for (int it = 0; it < 32; ++it) {
    const int kb = it * 32;
    __syncthreads();
#pragma unroll
    for (int i = 0; i < 2; ++i) {
      const int id = tid + i * 256;
      const int row = id >> 2, c = id & 3;
      cp16_g2s(Ain + (size_t)(mbase + row) * 1024 + kb + c * 8,
               &As[row * LDK + c * 8]);
    }
#pragma unroll
    for (int i = 0; i < 4; ++i) {
      const int id = tid + i * 256;
      const int n = id >> 2, c = id & 3;
      cp16_g2s(Wt0 + (size_t)(nbase + n) * 1024 + kb + c * 8,
               &Bs[n * LDK + c * 8]);
    }
    cp_join();
    __syncthreads();

    bf16x16 bfm[4];
#pragma unroll
    for (int nt = 0; nt < 4; ++nt) bfm[nt] = frag_ld(b_base[nt], k0);
#pragma unroll
    for (int mt = 0; mt < 4; ++mt) {
      bf16x16 af = frag_ld(a_base[mt], k0);
#pragma unroll
      for (int nt = 0; nt < 4; ++nt)
        acc[mt][nt] = WMMA_BF16(af, bfm[nt], acc[mt][nt]);
    }
  }

#pragma unroll
  for (int mt = 0; mt < 4; ++mt)
#pragma unroll
    for (int nt = 0; nt < 4; ++nt) {
      const int ng = nbase + wn * 64 + nt * 16 + lane16;
      const float bias = b0[ng];
#pragma unroll
      for (int r = 0; r < 8; ++r) {
        const int mg = mbase + wm * 64 + mt * 16 + r + half * 8;
        Out[(size_t)mg * 1024 + ng] = acc[mt][nt][r] + bias;
      }
    }
}

// ---------------------------------------------------------------------------
extern "C" void kernel_launch(void* const* d_in, const int* in_sizes, int n_in,
                              void* d_out, int out_size, void* d_ws,
                              size_t ws_size, hipStream_t stream) {
  (void)in_sizes; (void)n_in; (void)out_size; (void)ws_size;
  const float* X    = (const float*)d_in[0];
  const float* Wqkv = (const float*)d_in[1];
  const float* bqkv = (const float*)d_in[2];
  const float* W0   = (const float*)d_in[3];
  const float* b0   = (const float*)d_in[4];
  float* Out = (float*)d_out;

  // Workspace (bf16 elems): Q | K | V | Oattn | Xb | WtQKV | Wt0  (48 MB)
  const size_t HS  = (size_t)2 * 16 * 2048 * 64;  // 4194304
  const size_t XE  = (size_t)4096 * 1024;         // 4194304
  const size_t WQE = (size_t)3072 * 1024;         // 3145728
  bf16* Qb    = (bf16*)d_ws;
  bf16* Kb    = Qb + HS;
  bf16* Vb    = Kb + HS;
  bf16* Oattn = Vb + HS;
  bf16* Xb    = Oattn + HS;
  bf16* WtQ   = Xb + XE;
  bf16* Wt0   = WtQ + WQE;

  // Pre-pass: convert / transpose into L2-resident bf16 operands.
  mha_cvt_bf16<<<dim3(4096), 256, 0, stream>>>(X, Xb);
  mha_transpose_bf16<<<dim3(96, 32), 256, 0, stream>>>(Wqkv, WtQ, 1024, 3072);
  mha_transpose_bf16<<<dim3(32, 32), 256, 0, stream>>>(W0, Wt0, 1024, 1024);

  mha_qkv_gemm<<<dim3(12, 32), 256, 0, stream>>>(Xb, WtQ, bqkv, Qb, Kb, Vb);
  mha_flash_attn<<<dim3(16, 32), 256, 0, stream>>>(Qb, Kb, Vb, Oattn);
  mha_out_gemm<<<dim3(4, 32), 256, 0, stream>>>(Oattn, Wt0, b0, Out);
}
